// WaveNet_87402584473707
// MI455X (gfx1250) — matrix operand order repaired and checked by reading, not compiled, not run on hardware
//
#include <hip/hip_runtime.h>

#define DEVI __device__ __forceinline__

typedef __attribute__((ext_vector_type(16))) __bf16 v16bf;
typedef __attribute__((ext_vector_type(8)))  __bf16 v8bf;
typedef __attribute__((ext_vector_type(8)))  float  v8f;

constexpr int BATCH  = 16;
constexpr int TLEN   = 16384;
constexpr int CH     = 64;
constexpr int FIN    = 32;
constexpr int KW     = 3;
constexpr int NLAYER = 10;

// ---------------- WMMA tile load helpers ----------------
// A-matrix (16x32 bf16) element map: lane l, elem e ->
//   M = (l & 15), Klocal = e + (l>=16 ? 16 : 0)
DEVI v16bf load_a_lds(const __bf16* w, int mt, int s, int lane) {
  const int row  = mt * 16 + (lane & 15);
  const int lhi8 = (lane & 16) ? 16 : 0;
  const __bf16* p = w + row * 192 + s * 32 + lhi8;
  v8bf lo = *(const v8bf*)(p);        // Klocal = lhi8 .. lhi8+7
  v8bf hi = *(const v8bf*)(p + 8);    // Klocal = lhi8+8 .. lhi8+15
  return __builtin_shufflevector(lo, hi, 0,1,2,3,4,5,6,7,8,9,10,11,12,13,14,15);
}

// A tile from global f32 weights with row stride 64 (the 1x1 convs), on-the-fly bf16 convert
DEVI v16bf load_a_g(const float* w, int mt, int kh, int lane) {
  const int row  = mt * 16 + (lane & 15);
  const int lhi8 = (lane & 16) ? 16 : 0;
  const float* p = w + row * 64 + kh * 32 + lhi8;
  v16bf a;
#pragma unroll
  for (int e = 0; e < 8; ++e) a[e]     = (__bf16)p[e];
#pragma unroll
  for (int e = 0; e < 8; ++e) a[8 + e] = (__bf16)p[8 + e];
  return a;
}

// ---------------- input 1x1 conv + skip zero ----------------
__global__ __launch_bounds__(256) void wavenet_input(
    const float* __restrict__ x, const float* __restrict__ w_in,
    const float* __restrict__ b_in,
    unsigned short* __restrict__ h0_, float* __restrict__ skip) {
  __shared__ float wl[CH * FIN];
  __shared__ float bl[CH];
  for (int i = threadIdx.x; i < CH * FIN; i += 256) wl[i] = w_in[i];
  if (threadIdx.x < CH) bl[threadIdx.x] = b_in[threadIdx.x];
  __syncthreads();

  __bf16* h0 = (__bf16*)h0_;
  const size_t idx = (size_t)blockIdx.x * 256 + threadIdx.x;  // over B*T
  const float* xr = x + idx * FIN;
  float xv[FIN];
#pragma unroll
  for (int f = 0; f < FIN; ++f) xv[f] = xr[f];

  __bf16* hr = h0 + idx * CH;
#pragma unroll 4
  for (int c = 0; c < CH; ++c) {
    float acc = bl[c];
    const float* wr = wl + c * FIN;
#pragma unroll
    for (int f = 0; f < FIN; ++f) acc += wr[f] * xv[f];
    hr[c] = (__bf16)acc;
  }
  float4 z; z.x = z.y = z.z = z.w = 0.0f;
  float4* sr = (float4*)(skip + idx * CH);
#pragma unroll
  for (int q = 0; q < 16; ++q) sr[q] = z;
}

// ---------------- one WaveNet layer ----------------
__global__ __launch_bounds__(128) void wavenet_layer(
    const unsigned short* __restrict__ hin_, unsigned short* __restrict__ hout_,
    float* __restrict__ skip,
    const float* __restrict__ fw, const float* __restrict__ fb,
    const float* __restrict__ gw, const float* __restrict__ gb,
    const float* __restrict__ rw, const float* __restrict__ rb,
    const float* __restrict__ sw, const float* __restrict__ sb,
    int dil) {
  __shared__ __bf16 lfw[CH * 192];        // 24 KB  [cout][tap*64+cin]
  __shared__ __bf16 lgw[CH * 192];        // 24 KB
  __shared__ __bf16 stage[4 * 16 * CH];   //  8 KB  per-wave [n][c] staging

  const __bf16* hin = (const __bf16*)hin_;
  __bf16* hout = (__bf16*)hout_;
  const int tid = threadIdx.x;

  // weight fill: iterate global-linear (coalesced reads), scatter into LDS layout
  for (int i = tid; i < CH * CH * KW; i += 128) {
    int cout = i / (CH * KW);
    int rem  = i - cout * (CH * KW);
    int cin  = rem / KW;
    int tap  = rem - cin * KW;
    int li = cout * 192 + tap * 64 + cin;
    lfw[li] = (__bf16)fw[i];
    lgw[li] = (__bf16)gw[i];
  }
  __syncthreads();

  const int wave = tid >> 5;
  const int lane = tid & 31;
  const int nlo  = lane & 15;
  const int lhi  = lane >> 4;

  const int tiles_per_b = TLEN / 64;                      // 256
  const int b  = blockIdx.x / tiles_per_b;
  const int t0 = (blockIdx.x % tiles_per_b) * 64 + wave * 16;
  const int t  = t0 + nlo;                                // this lane's column
  const size_t rowbase = (size_t)b * TLEN;

  // ---- B tiles for the dilated conv (shared by filt & gate): 3 taps x 2 K-halves
  v16bf Bt[6];
#pragma unroll
  for (int tap = 0; tap < 3; ++tap) {
    const int ts = t - tap * dil;
    const __bf16* src = hin + (rowbase + (size_t)ts) * CH;
#pragma unroll
    for (int kh = 0; kh < 2; ++kh) {
      v16bf v;
      if (ts >= 0) {
        v = *(const v16bf*)(src + kh * 32 + lhi * 16);    // 32B contiguous channels
      } else {
#pragma unroll
        for (int e = 0; e < 16; ++e) v[e] = (__bf16)0.0f; // causal zero-pad
      }
      Bt[tap * 2 + kh] = v;
    }
  }

  // ---- accumulators seeded with biases (C/D map: M = mt*16 + lhi*8 + r)
  v8f facc[4], gacc[4];
#pragma unroll
  for (int mt = 0; mt < 4; ++mt)
#pragma unroll
    for (int r = 0; r < 8; ++r) {
      int c = mt * 16 + lhi * 8 + r;
      facc[mt][r] = fb[c];
      gacc[mt][r] = gb[c];
    }

  // ---- dilated convs: 4 M-tiles x 6 K-steps x 2 convs = 48 WMMAs
#pragma unroll
  for (int mt = 0; mt < 4; ++mt)
#pragma unroll
    for (int s = 0; s < 6; ++s) {
      v16bf af = load_a_lds(lfw, mt, s, lane);
      facc[mt] = __builtin_amdgcn_wmma_f32_16x16x32_bf16(
          false, af, false, Bt[s], (short)0, facc[mt], false, false);
      v16bf ag = load_a_lds(lgw, mt, s, lane);
      gacc[mt] = __builtin_amdgcn_wmma_f32_16x16x32_bf16(
          false, ag, false, Bt[s], (short)0, gacc[mt], false, false);
    }

  // ---- gated activation, stage C/D layout -> [n][c] bf16 in LDS (per-wave region)
  __bf16* stg = stage + wave * 16 * CH;
#pragma unroll
  for (int mt = 0; mt < 4; ++mt) {
    v8bf av;
#pragma unroll
    for (int r = 0; r < 8; ++r) {
      float fv = facc[mt][r];
      float gv = gacc[mt][r];
      float e2 = __expf(2.0f * fv);
      float th = __fdividef(e2 - 1.0f, e2 + 1.0f);        // tanh
      float sg = __fdividef(1.0f, 1.0f + __expf(-gv));    // sigmoid
      av[r] = (__bf16)(th * sg);
    }
    *(v8bf*)(stg + nlo * CH + mt * 16 + lhi * 8) = av;
  }
  // same-wave LDS ops are in-order: no barrier needed, region is wave-private

  // ---- act as B-matrix tiles (K = 64 -> two 32-halves)
  v16bf Ba[2];
#pragma unroll
  for (int kh = 0; kh < 2; ++kh)
    Ba[kh] = *(const v16bf*)(stg + nlo * CH + kh * 32 + lhi * 16);

  // ---- 1x1 residual & skip: 4 M-tiles x 2 K-halves x 2 = 16 WMMAs
  v8f racc[4], sacc[4];
#pragma unroll
  for (int mt = 0; mt < 4; ++mt)
#pragma unroll
    for (int r = 0; r < 8; ++r) {
      int c = mt * 16 + lhi * 8 + r;
      racc[mt][r] = rb[c];
      sacc[mt][r] = sb[c];
    }
#pragma unroll
  for (int mt = 0; mt < 4; ++mt)
#pragma unroll
    for (int kh = 0; kh < 2; ++kh) {
      v16bf ar = load_a_g(rw, mt, kh, lane);
      racc[mt] = __builtin_amdgcn_wmma_f32_16x16x32_bf16(
          false, ar, false, Ba[kh], (short)0, racc[mt], false, false);
      v16bf as = load_a_g(sw, mt, kh, lane);
      sacc[mt] = __builtin_amdgcn_wmma_f32_16x16x32_bf16(
          false, as, false, Ba[kh], (short)0, sacc[mt], false, false);
    }

  // ---- h_new = (h_old + residual) * 1/sqrt(2); restage for coalesced store
#pragma unroll
  for (int mt = 0; mt < 4; ++mt) {
    const v8bf ho = *(const v8bf*)(hin + (rowbase + (size_t)t) * CH + mt * 16 + lhi * 8);
    v8bf hv;
#pragma unroll
    for (int r = 0; r < 8; ++r)
      hv[r] = (__bf16)(((float)ho[r] + racc[mt][r]) * 0.70710678f);
    *(v8bf*)(stg + nlo * CH + mt * 16 + lhi * 8) = hv;
  }
  {
    // wave tile = 16 columns x 64 ch bf16 = 2KB, contiguous in [B][T][C]
    const uint4* s4 = (const uint4*)(stage + wave * 16 * CH);
    uint4* g4 = (uint4*)(hout + (rowbase + (size_t)t0) * CH);
#pragma unroll
    for (int i = 0; i < 4; ++i) g4[lane + i * 32] = s4[lane + i * 32];
  }

  // ---- skip accumulation: f32 RMW, 32B per lane
#pragma unroll
  for (int mt = 0; mt < 4; ++mt) {
    float* sp = skip + (rowbase + (size_t)t) * CH + mt * 16 + lhi * 8;
#pragma unroll
    for (int r = 0; r < 8; ++r) sp[r] += sacc[mt][r];
  }
}

// ---------------- output head ----------------
__global__ __launch_bounds__(256) void wavenet_output(
    const float* __restrict__ skip, const float* __restrict__ ow1,
    const float* __restrict__ ob1, const float* __restrict__ ow2,
    const float* __restrict__ ob2, float* __restrict__ out) {
  __shared__ float w1[CH * CH];
  __shared__ float w2[CH];
  __shared__ float b1[CH];
  for (int i = threadIdx.x; i < CH * CH; i += 256) w1[i] = ow1[i];
  if (threadIdx.x < CH) { w2[threadIdx.x] = ow2[threadIdx.x]; b1[threadIdx.x] = ob1[threadIdx.x]; }
  __syncthreads();

  const size_t idx = (size_t)blockIdx.x * 256 + threadIdx.x;
  const float* sr = skip + idx * CH;
  float s[CH];
#pragma unroll
  for (int c = 0; c < CH; ++c) s[c] = fmaxf(sr[c], 0.0f);

  float y = ob2[0];
  for (int o = 0; o < CH; ++o) {
    float acc = b1[o];
    const float4* wr = (const float4*)(w1 + o * CH);
#pragma unroll
    for (int q = 0; q < 16; ++q) {
      float4 wv = wr[q];
      acc += wv.x * s[q * 4] + wv.y * s[q * 4 + 1] + wv.z * s[q * 4 + 2] + wv.w * s[q * 4 + 3];
    }
    y += w2[o] * fmaxf(acc, 0.0f);
  }
  out[idx] = y;
}

// ---------------- launch ----------------
extern "C" void kernel_launch(void* const* d_in, const int* in_sizes, int n_in,
                              void* d_out, int out_size, void* d_ws, size_t ws_size,
                              hipStream_t stream) {
  (void)in_sizes; (void)n_in; (void)out_size; (void)ws_size;
  const float* x    = (const float*)d_in[0];
  const float* w_in = (const float*)d_in[1];
  const float* b_in = (const float*)d_in[2];
  const float* fw   = (const float*)d_in[3];
  const float* fb   = (const float*)d_in[4];
  const float* gw   = (const float*)d_in[5];
  const float* gb   = (const float*)d_in[6];
  const float* rw   = (const float*)d_in[7];
  const float* rb   = (const float*)d_in[8];
  const float* sw   = (const float*)d_in[9];
  const float* sb   = (const float*)d_in[10];
  const float* ow1  = (const float*)d_in[11];
  const float* ob1  = (const float*)d_in[12];
  const float* ow2  = (const float*)d_in[13];
  const float* ob2  = (const float*)d_in[14];
  float* out = (float*)d_out;

  const size_t HBYTES = (size_t)BATCH * TLEN * CH * 2;   // bf16 h buffer: 32 MB
  char* ws = (char*)d_ws;
  unsigned short* hA = (unsigned short*)ws;
  unsigned short* hB = (unsigned short*)(ws + HBYTES);
  float* skip = (float*)(ws + 2 * HBYTES);               // 64 MB f32

  wavenet_input<<<BATCH * TLEN / 256, 256, 0, stream>>>(x, w_in, b_in, hA, skip);

  const unsigned short* hin = hA;
  unsigned short* hout = hB;
  for (int i = 0; i < NLAYER; ++i) {
    int dil = 1 << (i % 10);
    wavenet_layer<<<BATCH * (TLEN / 64), 128, 0, stream>>>(
        hin, hout, skip,
        fw + (size_t)i * CH * CH * KW, fb + (size_t)i * CH,
        gw + (size_t)i * CH * CH * KW, gb + (size_t)i * CH,
        rw + (size_t)i * CH * CH,      rb + (size_t)i * CH,
        sw + (size_t)i * CH * CH,      sb + (size_t)i * CH,
        dil);
    unsigned short* tmp = (unsigned short*)hin;
    hin = hout;
    hout = tmp;
  }

  wavenet_output<<<BATCH * TLEN / 256, 256, 0, stream>>>(skip, ow1, ob1, ow2, ob2, out);
}
